// MoE_21337397526571
// MI455X (gfx1250) — compile-verified, matrix operand order checked by reading, not run on hardware
//
#include <hip/hip_runtime.h>
#include <hip/hip_bf16.h>
#include <stdint.h>

// Problem sizes (match reference)
#define B_TOK  4096
#define D_IN   1024
#define H_DIM  4096
#define E_NUM  8
#define GH_DIM 512
#define O_DIM  1024

typedef __attribute__((ext_vector_type(16))) __bf16 v16bf;
typedef __attribute__((ext_vector_type(8)))  __bf16 v8bf;
typedef __attribute__((ext_vector_type(8)))  float  v8f;

// f32 pair -> packed bf16 (RNE) in one VALU op via the gfx1250 packed convert.
// (clang has no builtin for the 2-source form here; inline asm emits it directly)
__device__ __forceinline__ uint32_t pack_bf16(float a, float b) {
  uint32_t u;
  asm("v_cvt_pk_bf16_f32 %0, %1, %2" : "=v"(u) : "v"(a), "v"(b));
  return u;
}

union FragU { struct { v8bf lo, hi; } p; v16bf v; };

// LDS tile: 128 rows x 32 K bf16, row stride 40 uint16 (80B, 16B-aligned, bank-spread)
// Fragment layout per ISA 7.12.2 (16-bit A 16x32): lane l%16 = row, h = l/16:
//   VGPR0..3 = K in [8h, 8h+8)  -> u16 idx  8h
//   VGPR4..7 = K in [16+8h, +8) -> u16 idx 16+8h
__device__ __forceinline__ v16bf load_frag(const uint16_t* base, int row, int h) {
  FragU u;
  const uint16_t* r = base + row * 40;
  u.p.lo = *reinterpret_cast<const v8bf*>(r + h * 8);
  u.p.hi = *reinterpret_cast<const v8bf*>(r + 16 + h * 8);
  return u.v;
}

// C[M,N] = act(A[M,K] @ W[N,K]^T + bias[N]); M,N multiples of 128, K multiple of 32.
// Block 256 threads = 8 wave32s; 128x128 C tile; wave grid 4(M) x 2(N); wave tile 32x64.
__global__ __launch_bounds__(256) void gemm_bf16_wmma(
    const float* __restrict__ A, const float* __restrict__ W,
    const float* __restrict__ bias, float* __restrict__ C,
    int K, int ldA, int ldW, int ldC, int do_relu)
{
  __shared__ __align__(16) uint16_t ldsA[2][128 * 40];
  __shared__ __align__(16) uint16_t ldsB[2][128 * 40];

  const int tid  = threadIdx.x;
  const int lane = tid & 31;
  const int wave = tid >> 5;
  const int wr   = wave >> 1;   // 0..3  (M direction)
  const int wc   = wave & 1;    // 0..1  (N direction)
  const int lr   = lane & 15;
  const int lh   = lane >> 4;

  const int blockM = blockIdx.y * 128;
  const int blockN = blockIdx.x * 128;

  // staging: thread t loads half-row (16 f32) of A-tile row t/2 and W-tile row t/2
  const int srow  = tid >> 1;
  const int shalf = tid & 1;
  const float* gA = A + (size_t)(blockM + srow) * ldA + shalf * 16;
  const float* gW = W + (size_t)(blockN + srow) * ldW + shalf * 16;

  const int nk = K >> 5;

  v8f acc[2][4];
  #pragma unroll
  for (int i = 0; i < 2; ++i)
    #pragma unroll
    for (int j = 0; j < 4; ++j)
      acc[i][j] = v8f{0.f,0.f,0.f,0.f,0.f,0.f,0.f,0.f};

  float4 rA[4], rW[4];
  {
    const float4* pA = reinterpret_cast<const float4*>(gA);
    const float4* pW = reinterpret_cast<const float4*>(gW);
    #pragma unroll
    for (int c = 0; c < 4; ++c) { rA[c] = pA[c]; rW[c] = pW[c]; }
  }

  for (int kt = 0; kt < nk; ++kt) {
    const int buf = kt & 1;

    // convert current tile to bf16, store to LDS (2x ds_store_b128 per operand)
    uint16_t* sA = ldsA[buf] + srow * 40 + shalf * 16;
    uint16_t* sB = ldsB[buf] + srow * 40 + shalf * 16;
    uint32_t pa[8], pb[8];
    #pragma unroll
    for (int c = 0; c < 4; ++c) {
      pa[2*c]   = pack_bf16(rA[c].x, rA[c].y);
      pa[2*c+1] = pack_bf16(rA[c].z, rA[c].w);
      pb[2*c]   = pack_bf16(rW[c].x, rW[c].y);
      pb[2*c+1] = pack_bf16(rW[c].z, rW[c].w);
    }
    *reinterpret_cast<uint4*>(sA)     = make_uint4(pa[0], pa[1], pa[2], pa[3]);
    *reinterpret_cast<uint4*>(sA + 8) = make_uint4(pa[4], pa[5], pa[6], pa[7]);
    *reinterpret_cast<uint4*>(sB)     = make_uint4(pb[0], pb[1], pb[2], pb[3]);
    *reinterpret_cast<uint4*>(sB + 8) = make_uint4(pb[4], pb[5], pb[6], pb[7]);

    // issue next tile's global loads before the barrier (overlap with WMMA below)
    if (kt + 1 < nk) {
      const float4* pA = reinterpret_cast<const float4*>(gA + (size_t)(kt + 1) * 32);
      const float4* pW = reinterpret_cast<const float4*>(gW + (size_t)(kt + 1) * 32);
      #pragma unroll
      for (int c = 0; c < 4; ++c) { rA[c] = pA[c]; rW[c] = pW[c]; }
      if (kt + 2 < nk) {
        __builtin_prefetch(gA + (size_t)(kt + 2) * 32, 0, 3);
        __builtin_prefetch(gW + (size_t)(kt + 2) * 32, 0, 3);
      }
    }
    __syncthreads();

    v16bf aF[2], bF[4];
    #pragma unroll
    for (int i = 0; i < 2; ++i)
      aF[i] = load_frag(ldsA[buf], wr * 32 + i * 16 + lr, lh);
    #pragma unroll
    for (int j = 0; j < 4; ++j)
      bF[j] = load_frag(ldsB[buf], wc * 64 + j * 16 + lr, lh);

    #pragma unroll
    for (int i = 0; i < 2; ++i)
      #pragma unroll
      for (int j = 0; j < 4; ++j)
        acc[i][j] = __builtin_amdgcn_wmma_f32_16x16x32_bf16(
            false, aF[i], false, bF[j], (short)0, acc[i][j], false, false);

    __syncthreads();
  }

  // epilogue: C/D layout — VGPR v: lanes 0-15 M=v, lanes 16-31 M=v+8; N = lane%16
  float bj[4];
  #pragma unroll
  for (int j = 0; j < 4; ++j)
    bj[j] = bias[blockN + wc * 64 + j * 16 + lr];

  #pragma unroll
  for (int i = 0; i < 2; ++i) {
    #pragma unroll
    for (int j = 0; j < 4; ++j) {
      const int n = blockN + wc * 64 + j * 16 + lr;
      #pragma unroll
      for (int v = 0; v < 8; ++v) {
        const int m = blockM + wr * 32 + i * 16 + lh * 8 + v;
        float val = acc[i][j][v] + bj[j];
        if (do_relu) val = fmaxf(val, 0.f);
        C[(size_t)m * ldC + n] = val;
      }
    }
  }
}

// logits[b,e] = gh[b,:] . gw2[e,:] + gb2[e]   (B*E threads, K=512)
__global__ void gate_logits_kernel(const float* __restrict__ gh,
                                   const float* __restrict__ gw2,
                                   const float* __restrict__ gb2,
                                   float* __restrict__ logits)
{
  int g = blockIdx.x * blockDim.x + threadIdx.x;
  if (g >= B_TOK * E_NUM) return;
  int b = g >> 3, e = g & 7;
  const float4* a = reinterpret_cast<const float4*>(gh + (size_t)b * GH_DIM);
  const float4* w = reinterpret_cast<const float4*>(gw2 + (size_t)e * GH_DIM);
  float s = 0.f;
  #pragma unroll 4
  for (int i = 0; i < GH_DIM / 4; ++i) {
    float4 av = a[i], wv = w[i];
    s += av.x * wv.x + av.y * wv.y + av.z * wv.z + av.w * wv.w;
  }
  logits[g] = s + gb2[e];
}

// softmax over E=8 + deterministic categorical sample (hash RNG, stand-in for key=42)
__global__ void gate_softmax_sample_kernel(const float* __restrict__ logits,
                                           float* __restrict__ gate_out,
                                           float* __restrict__ idx_f,
                                           int* __restrict__ idx_i)
{
  int b = blockIdx.x * blockDim.x + threadIdx.x;
  if (b >= B_TOK) return;
  float l[E_NUM];
  float m = -1e30f;
  #pragma unroll
  for (int e = 0; e < E_NUM; ++e) { l[e] = logits[b * E_NUM + e]; m = fmaxf(m, l[e]); }
  float s = 0.f;
  #pragma unroll
  for (int e = 0; e < E_NUM; ++e) { l[e] = __expf(l[e] - m); s += l[e]; }
  float inv = 1.f / s;

  uint32_t h = (uint32_t)b * 747796405u + 2891336453u + 42u;
  h ^= h >> 17; h *= 0xED5AD4BBu; h ^= h >> 11; h *= 0xAC4C1B51u;
  h ^= h >> 15; h *= 0x31848BABu; h ^= h >> 14;
  float u = (float)(h >> 8) * (1.0f / 16777216.0f);

  float cdf = 0.f;
  int idx = -1;
  #pragma unroll
  for (int e = 0; e < E_NUM; ++e) {
    float p = l[e] * inv;
    gate_out[b * E_NUM + e] = p;
    cdf += p;
    if (idx < 0 && u < cdf) idx = e;
  }
  if (idx < 0) idx = E_NUM - 1;
  idx_f[b] = (float)idx;
  idx_i[b] = idx;
}

// in-place row softmax over O=1024; one 256-thread block per row
__global__ __launch_bounds__(256) void softmax_row_kernel(float* __restrict__ data)
{
  __shared__ float red[256];
  float* p = data + (size_t)blockIdx.x * O_DIM;
  const int t = threadIdx.x;
  float4 v = reinterpret_cast<float4*>(p)[t];

  float m = fmaxf(fmaxf(v.x, v.y), fmaxf(v.z, v.w));
  red[t] = m; __syncthreads();
  for (int s = 128; s > 0; s >>= 1) {
    if (t < s) red[t] = fmaxf(red[t], red[t + s]);
    __syncthreads();
  }
  m = red[0]; __syncthreads();

  v.x = __expf(v.x - m); v.y = __expf(v.y - m);
  v.z = __expf(v.z - m); v.w = __expf(v.w - m);
  red[t] = v.x + v.y + v.z + v.w; __syncthreads();
  for (int s = 128; s > 0; s >>= 1) {
    if (t < s) red[t] += red[t + s];
    __syncthreads();
  }
  float inv = 1.f / red[0];
  v.x *= inv; v.y *= inv; v.z *= inv; v.w *= inv;
  reinterpret_cast<float4*>(p)[t] = v;
}

// final[b,:] = expert_out[b, idx[b], :]
__global__ void gather_final_kernel(const float* __restrict__ eout,
                                    const int* __restrict__ idx_i,
                                    float* __restrict__ final_out)
{
  const int b = blockIdx.x;
  const int e = idx_i[b];
  const float4* src = reinterpret_cast<const float4*>(eout + ((size_t)b * E_NUM + e) * O_DIM);
  float4* dst = reinterpret_cast<float4*>(final_out + (size_t)b * O_DIM);
  dst[threadIdx.x] = src[threadIdx.x];
}

extern "C" void kernel_launch(void* const* d_in, const int* in_sizes, int n_in,
                              void* d_out, int out_size, void* d_ws, size_t ws_size,
                              hipStream_t stream)
{
  (void)in_sizes; (void)n_in; (void)out_size; (void)ws_size;
  const float* x   = (const float*)d_in[0];
  const float* gw1 = (const float*)d_in[1];
  const float* gb1 = (const float*)d_in[2];
  const float* gw2 = (const float*)d_in[3];
  const float* gb2 = (const float*)d_in[4];
  const float* ew1 = (const float*)d_in[5];
  const float* eb1 = (const float*)d_in[6];
  const float* ew2 = (const float*)d_in[7];
  const float* eb2 = (const float*)d_in[8];

  // d_out: final [B,O] | expert_out [B,E,O] | gate [B,E] | idx [B]
  float* out       = (float*)d_out;
  float* final_out = out;
  float* eout      = out + (size_t)B_TOK * O_DIM;
  float* gate      = eout + (size_t)B_TOK * E_NUM * O_DIM;
  float* idxf      = gate + (size_t)B_TOK * E_NUM;

  // workspace (~84 MB): gh [B,GH] | logits [B,E] | idx int [B] | eh [B,H] (per-expert reuse)
  float* ws     = (float*)d_ws;
  float* gh     = ws;
  float* logits = gh + (size_t)B_TOK * GH_DIM;
  int*   idxi   = (int*)(logits + (size_t)B_TOK * E_NUM);
  float* eh     = ws + (size_t)4 * 1024 * 1024;  // 16MB mark, past gh/logits/idx

  dim3 blk(256);

  // gating: gh = relu(x @ gw1^T + gb1)   [4096x512], K=1024
  gemm_bf16_wmma<<<dim3(GH_DIM / 128, B_TOK / 128), blk, 0, stream>>>(
      x, gw1, gb1, gh, D_IN, D_IN, D_IN, GH_DIM, 1);
  gate_logits_kernel<<<(B_TOK * E_NUM) / 256, blk, 0, stream>>>(gh, gw2, gb2, logits);
  gate_softmax_sample_kernel<<<B_TOK / 256, blk, 0, stream>>>(logits, gate, idxf, idxi);

  for (int e = 0; e < E_NUM; ++e) {
    const float* w1 = ew1 + (size_t)e * H_DIM * D_IN;
    const float* b1 = eb1 + (size_t)e * H_DIM;
    const float* w2 = ew2 + (size_t)e * O_DIM * H_DIM;
    const float* b2 = eb2 + (size_t)e * O_DIM;
    // eh = relu(x @ ew1[e]^T + eb1[e])   [4096x4096], K=1024
    gemm_bf16_wmma<<<dim3(H_DIM / 128, B_TOK / 128), blk, 0, stream>>>(
        x, w1, b1, eh, D_IN, D_IN, D_IN, H_DIM, 1);
    // expert_out[:,e,:] = eh @ ew2[e]^T + eb2[e]   [4096x1024], K=4096, ldC = E*O
    gemm_bf16_wmma<<<dim3(O_DIM / 128, B_TOK / 128), blk, 0, stream>>>(
        eh, w2, b2, eout + (size_t)e * O_DIM, H_DIM, H_DIM, H_DIM, E_NUM * O_DIM, 0);
  }

  softmax_row_kernel<<<B_TOK * E_NUM, blk, 0, stream>>>(eout);
  gather_final_kernel<<<B_TOK, blk, 0, stream>>>(eout, idxi, final_out);
}
